// BitLinear_16716012716233
// MI455X (gfx1250) — 8-bit matrix operand order repaired and checked by reading, compiled, not run on hardware
//
#include <hip/hip_runtime.h>
#include <cstdint>
#include <cstddef>


typedef int   v8i  __attribute__((ext_vector_type(8)));
typedef int   v16i __attribute__((ext_vector_type(16)));
typedef float v8f  __attribute__((ext_vector_type(8)));
typedef int   v4ib __attribute__((vector_size(16)));

#define EPSQ 1e-5f

#if __has_builtin(__builtin_amdgcn_global_load_async_to_lds_b128)
#define USE_ASYNC_LDS 1
#else
#define USE_ASYNC_LDS 0
#endif

__device__ __forceinline__ void wait_asynccnt0() {
#if __has_builtin(__builtin_amdgcn_s_wait_asynccnt)
    __builtin_amdgcn_s_wait_asynccnt(0);
#else
    asm volatile("s_wait_asynccnt 0x0" ::: "memory");
#endif
}

#if USE_ASYNC_LDS
__device__ __forceinline__ void async_cp16(const uint8_t* g, uint8_t* l) {
    __builtin_amdgcn_global_load_async_to_lds_b128(
        (__attribute__((address_space(1))) v4ib*)(void*)g,
        (__attribute__((address_space(3))) v4ib*)(void*)l,
         0,  0);
}
#endif

__device__ static const uint8_t FP8_LUT[16] = {
    0xD0, 0xCE, 0xCC, 0xCA, 0xC8, 0xC4, 0xC0, 0xB8,
    0x00, 0x38, 0x40, 0x44, 0x48, 0x4A, 0x4C, 0x4E
};

static constexpr size_t NK      = 16384ull * 4096ull;
static constexpr size_t MK      = 8192ull * 4096ull;
static constexpr size_t WQ_OFF  = 0;
static constexpr size_t QX_OFF  = WQ_OFF + NK;
static constexpr size_t XS_OFF  = QX_OFF + MK;
static constexpr size_t PART_OFF= XS_OFF + 8192ull * 4;
static constexpr size_t WSCL_OFF= PART_OFF + 4096ull * 4;

__global__ __launch_bounds__(256) void k_wabs_partial(
    const float* __restrict__ w, float* __restrict__ part)
{
    __shared__ float red[256];
    const size_t base = (size_t)blockIdx.x * 16384;
    float s = 0.f;
    #pragma unroll 4
    for (int i = 0; i < 64; ++i)
        s += fabsf(w[base + threadIdx.x + (size_t)i * 256]);
    red[threadIdx.x] = s;
    __syncthreads();
    for (int off = 128; off > 0; off >>= 1) {
        if (threadIdx.x < (unsigned)off)
            red[threadIdx.x] += red[threadIdx.x + off];
        __syncthreads();
    }
    if (threadIdx.x == 0) part[blockIdx.x] = red[0];
}

__global__ __launch_bounds__(256) void k_wscale_final(
    const float* __restrict__ part, float* __restrict__ wscl)
{
    __shared__ float red[256];
    float s = 0.f;
    for (int i = threadIdx.x; i < 4096; i += 256) s += part[i];
    red[threadIdx.x] = s;
    __syncthreads();
    for (int off = 128; off > 0; off >>= 1) {
        if (threadIdx.x < (unsigned)off)
            red[threadIdx.x] += red[threadIdx.x + off];
        __syncthreads();
    }
    if (threadIdx.x == 0)
        wscl[0] = fmaxf(red[0] * (1.0f / (float)NK), EPSQ);
}

__global__ __launch_bounds__(256) void k_wquant(
    const float* __restrict__ w, uint8_t* __restrict__ wq,
    const float* __restrict__ wscl)
{
    const float inv = 1.0f / wscl[0];
    const size_t i0 = ((size_t)blockIdx.x * 256 + threadIdx.x) * 4;
    const float4 v = *(const float4*)(w + i0);
    uchar4 o;
    o.x = FP8_LUT[(int)fminf(fmaxf(rintf(v.x * inv), -1.f), 1.f) + 8];
    o.y = FP8_LUT[(int)fminf(fmaxf(rintf(v.y * inv), -1.f), 1.f) + 8];
    o.z = FP8_LUT[(int)fminf(fmaxf(rintf(v.z * inv), -1.f), 1.f) + 8];
    o.w = FP8_LUT[(int)fminf(fmaxf(rintf(v.w * inv), -1.f), 1.f) + 8];
    *(uchar4*)(wq + i0) = o;
}

__global__ __launch_bounds__(256) void k_actquant(
    const float* __restrict__ x, uint8_t* __restrict__ qx,
    float* __restrict__ xs, int K)
{
    __shared__ float red[256];
    const size_t row = blockIdx.x;
    const float* xr = x + row * (size_t)K;
    float amax = 0.f;
    for (int k = threadIdx.x; k < K; k += 256)
        amax = fmaxf(amax, fabsf(xr[k]));
    red[threadIdx.x] = amax;
    __syncthreads();
    for (int off = 128; off > 0; off >>= 1) {
        if (threadIdx.x < (unsigned)off)
            red[threadIdx.x] = fmaxf(red[threadIdx.x], red[threadIdx.x + off]);
        __syncthreads();
    }
    const float s   = fmaxf(red[0], EPSQ) * (1.0f / 7.0f);
    const float inv = 1.0f / s;
    uint8_t* qr = qx + row * (size_t)K;
    for (int k = threadIdx.x; k < K; k += 256) {
        const int q = (int)fminf(fmaxf(rintf(xr[k] * inv), -8.f), 7.f);
        qr[k] = FP8_LUT[q + 8];
    }
    if (threadIdx.x == 0) xs[row] = s;
}

#define TILE_M 128
#define TILE_N 128
#define TILE_K 128
#define LDSS   144

__global__ __launch_bounds__(256) void k_gemm_fp8(
    const uint8_t* __restrict__ qx,
    const uint8_t* __restrict__ wq,
    const float*   __restrict__ xs,
    const float*   __restrict__ wscl,
    float* __restrict__ out,
    int M, int N, int K)
{
    __shared__ uint8_t lA[2][TILE_M * LDSS];
    __shared__ uint8_t lB[2][TILE_N * LDSS];

    const int tid  = threadIdx.x;
    const int lane = tid & 31;
    const int wave = tid >> 5;
    const int wm   = wave >> 2;
    const int wn   = wave & 3;
    const int g    = lane >> 4;
    const int lm   = lane & 15;

    const int bm = blockIdx.y * TILE_M;
    const int bn = blockIdx.x * TILE_N;

    const int sr = tid >> 1;
    const int sc = (tid & 1) << 4;
    const uint8_t* gA = qx + (size_t)(bm + sr) * K + sc;
    const uint8_t* gB = wq + (size_t)(bn + sr) * K + sc;
    const int lo = sr * LDSS + sc;

    v8f acc[4][2];
    const v8f fzero = {};
    #pragma unroll
    for (int mt = 0; mt < 4; ++mt)
        #pragma unroll
        for (int nt = 0; nt < 2; ++nt)
            acc[mt][nt] = fzero;

    const int NIT = K / TILE_K;

#if USE_ASYNC_LDS
    #pragma unroll
    for (int c = 0; c < 4; ++c) {
        async_cp16(gA + 32 * c, &lA[0][lo + 32 * c]);
        async_cp16(gB + 32 * c, &lB[0][lo + 32 * c]);
    }

    for (int it = 0; it < NIT; ++it) {
        const int buf = it & 1;
        wait_asynccnt0();
        __syncthreads();
        if (it + 1 < NIT) {
            const int nk = (it + 1) * TILE_K;
            const int nb = buf ^ 1;
            #pragma unroll
            for (int c = 0; c < 4; ++c) {
                async_cp16(gA + nk + 32 * c, &lA[nb][lo + 32 * c]);
                async_cp16(gB + nk + 32 * c, &lB[nb][lo + 32 * c]);
            }
        }
#else
    int4 ra[4], rb[4];
    #pragma unroll
    for (int c = 0; c < 4; ++c) {
        ra[c] = *(const int4*)(gA + 32 * c);
        rb[c] = *(const int4*)(gB + 32 * c);
    }

    for (int it = 0; it < NIT; ++it) {
        const int buf = it & 1;
        #pragma unroll
        for (int c = 0; c < 4; ++c) {
            *(int4*)(&lA[buf][lo + 32 * c]) = ra[c];
            *(int4*)(&lB[buf][lo + 32 * c]) = rb[c];
        }
        if (it + 1 < NIT) {
            const int nk = (it + 1) * TILE_K;
            #pragma unroll
            for (int c = 0; c < 4; ++c) {
                ra[c] = *(const int4*)(gA + nk + 32 * c);
                rb[c] = *(const int4*)(gB + nk + 32 * c);
            }
        }
        __syncthreads();
#endif
        v16i afrag[4];
        #pragma unroll
        for (int mt = 0; mt < 4; ++mt) {
            const uint8_t* base = &lA[buf][(wm * 64 + mt * 16 + lm) * LDSS];
            #pragma unroll
            for (int j = 0; j < 16; ++j) {
                const int jl = j & 7;
                const int kb = (j >> 2) * 32 + g * 16 + (j & 3) * 4;
                afrag[mt][j] = *(const int*)(base + kb);
            }
        }
        #pragma unroll
        for (int nt = 0; nt < 2; ++nt) {
            const uint8_t* base = &lB[buf][(wn * 32 + nt * 16 + lm) * LDSS];
            v16i bfrag;
            #pragma unroll
            for (int j = 0; j < 16; ++j) {
                const int kb = (j >> 2) * 32 + g * 16 + (j & 3) * 4;
                bfrag[j] = *(const int*)(base + kb);
            }
            #pragma unroll
            for (int mt = 0; mt < 4; ++mt) {
                acc[mt][nt] = __builtin_amdgcn_wmma_f32_16x16x128_fp8_fp8(
                    afrag[mt], bfrag, (short)0, acc[mt][nt],
                     false,  false);
            }
        }
    }

    const float sw = wscl[0];
    #pragma unroll
    for (int mt = 0; mt < 4; ++mt) {
        const int mbase = bm + wm * 64 + mt * 16 + g * 8;
        #pragma unroll
        for (int nt = 0; nt < 2; ++nt) {
            const int ncol = bn + wn * 32 + nt * 16 + lm;
            #pragma unroll
            for (int r = 0; r < 8; ++r) {
                const int mrow = mbase + r;
                out[(size_t)mrow * N + ncol] =
                    acc[mt][nt][r] * xs[mrow] * sw;
            }
        }
    }
}

extern "C" void kernel_launch(void* const* d_in, const int* in_sizes, int n_in,
                              void* d_out, int out_size, void* d_ws, size_t ws_size,
                              hipStream_t stream)
{
    const float* x = (const float*)d_in[0];
    const float* w = (const float*)d_in[1];

    const int K = 4096;
    const int M = in_sizes[0] / K;
    const int N = in_sizes[1] / K;

    uint8_t* ws   = (uint8_t*)d_ws;
    uint8_t* wq   = ws + WQ_OFF;
    uint8_t* qx   = ws + QX_OFF;
    float*   xsc  = (float*)(ws + XS_OFF);
    float*   part = (float*)(ws + PART_OFF);
    float*   wscl = (float*)(ws + WSCL_OFF);
    float*   out  = (float*)d_out;

    k_wabs_partial<<<4096, 256, 0, stream>>>(w, part);
    k_wscale_final<<<1, 256, 0, stream>>>(part, wscl);
    k_wquant<<<(int)(NK / (256ull * 4ull)), 256, 0, stream>>>(w, wq, wscl);
    k_actquant<<<M, 256, 0, stream>>>(x, qx, xsc, K);

    dim3 grid(N / TILE_N, M / TILE_M);
    k_gemm_fp8<<<grid, 256, 0, stream>>>(qx, wq, xsc, wscl, out, M, N, K);
    (void)n_in; (void)out_size; (void)ws_size;
}
